// TransformerDecoder_28338194219740
// MI455X (gfx1250) — matrix operand order repaired and checked by reading, not compiled, not run on hardware
//
#include <hip/hip_runtime.h>

// ---------------------------------------------------------------------------
// Types
// ---------------------------------------------------------------------------
typedef __bf16 bf16;
typedef __attribute__((ext_vector_type(16))) __bf16 bf16x16;
typedef __attribute__((ext_vector_type(8)))  __bf16 bf16x8;
typedef __attribute__((ext_vector_type(4)))  __bf16 bf16x4;
typedef __attribute__((ext_vector_type(8)))  float  f32x8;
typedef int v4i_ __attribute__((vector_size(16)));

// Model dims
#define B_     8
#define N_     512
#define S_     512
#define DIM_   512
#define HEADS_ 8
#define DHEAD_ 512
#define INNER_ 4096
#define MLP_   2048
#define TOK_   (B_ * N_)          // 4096 tokens
#define SCALE_ 0.04419417382415922f  // 512^-0.5

// GEMM epilogue modes
#define MODE_BF16          0   // D -> bf16
#define MODE_SCALE_F32     1   // D*scale -> f32
#define MODE_BIAS_GELU_B16 2   // gelu(D + bias) -> bf16
#define MODE_BIAS_RES_F32  3   // D + bias + res -> f32

// CDNA5 async global->LDS DMA availability (device pass only)
#if defined(__gfx1250__) && __has_builtin(__builtin_amdgcn_global_load_async_to_lds_b128) && __has_builtin(__builtin_amdgcn_s_wait_asynccnt)
#define USE_ASYNC_LDS 1
#else
#define USE_ASYNC_LDS 0
#endif

// ---------------------------------------------------------------------------
// Helpers
// ---------------------------------------------------------------------------
__device__ __forceinline__ float gelu_exact(float x) {
    return 0.5f * x * (1.0f + erff(x * 0.70710678118654752440f));
}

__device__ __forceinline__ bf16x16 cat8(bf16x8 lo, bf16x8 hi) {
    return __builtin_shufflevector(lo, hi, 0,1,2,3,4,5,6,7,8,9,10,11,12,13,14,15);
}

#if USE_ASYNC_LDS
// one 16-byte async DMA: LDS[l] = GLOBAL[g]   (tracked by ASYNCcnt)
__device__ __forceinline__ void async_cp16(const bf16* g, bf16* l) {
    __builtin_amdgcn_global_load_async_to_lds_b128(
        (__attribute__((address_space(1))) v4i_*)(void*)g,
        (__attribute__((address_space(3))) v4i_*)(void*)l,
        0, 0);
}
#endif

// ---------------------------------------------------------------------------
// fp32 -> bf16 conversion (vectorized x4, grid-stride)
// ---------------------------------------------------------------------------
__global__ void cvt_f32_bf16(const float4* __restrict__ src, bf16* __restrict__ dst, long n4) {
    long i = (long)blockIdx.x * blockDim.x + threadIdx.x;
    long stride = (long)gridDim.x * blockDim.x;
    for (; i < n4; i += stride) {
        float4 f = src[i];
        bf16x4 v = { (__bf16)f.x, (__bf16)f.y, (__bf16)f.z, (__bf16)f.w };
        *(bf16x4*)(dst + 4 * i) = v;
    }
}

// ---------------------------------------------------------------------------
// LayerNorm over last dim (512), one wave32 per row, fp32 in -> bf16 out
// ---------------------------------------------------------------------------
__global__ __launch_bounds__(256) void layernorm_bf16(
    const float* __restrict__ x, const float* __restrict__ w,
    const float* __restrict__ b, bf16* __restrict__ out, int nrows) {
    int wid = blockIdx.x * 8 + (threadIdx.x >> 5);
    if (wid >= nrows) return;
    int lane = threadIdx.x & 31;
    const float* row = x + (size_t)wid * DIM_;
    float v[16];
    float s = 0.f;
#pragma unroll
    for (int i = 0; i < 16; i++) { v[i] = row[lane + 32 * i]; s += v[i]; }
#pragma unroll
    for (int o = 16; o > 0; o >>= 1) s += __shfl_xor(s, o, 32);
    float mean = s * (1.0f / DIM_);
    float var = 0.f;
#pragma unroll
    for (int i = 0; i < 16; i++) { float d = v[i] - mean; var += d * d; }
#pragma unroll
    for (int o = 16; o > 0; o >>= 1) var += __shfl_xor(var, o, 32);
    float inv = rsqrtf(var * (1.0f / DIM_) + 1e-5f);
    bf16* orow = out + (size_t)wid * DIM_;
#pragma unroll
    for (int i = 0; i < 16; i++) {
        int c = lane + 32 * i;
        orow[c] = (__bf16)((v[i] - mean) * inv * w[c] + b[c]);
    }
}

// ---------------------------------------------------------------------------
// Row softmax over 512 cols, one wave32 per row, f32 (pre-scaled) -> bf16
// ---------------------------------------------------------------------------
__global__ __launch_bounds__(256) void softmax_bf16(
    const float* __restrict__ s, bf16* __restrict__ p, int nrows) {
    int wid = blockIdx.x * 8 + (threadIdx.x >> 5);
    if (wid >= nrows) return;
    int lane = threadIdx.x & 31;
    const float* row = s + (size_t)wid * 512;
    float v[16];
    float m = -3.4e38f;
#pragma unroll
    for (int i = 0; i < 16; i++) { v[i] = row[lane + 32 * i]; m = fmaxf(m, v[i]); }
#pragma unroll
    for (int o = 16; o > 0; o >>= 1) m = fmaxf(m, __shfl_xor(m, o, 32));
    float sum = 0.f;
#pragma unroll
    for (int i = 0; i < 16; i++) { v[i] = __expf(v[i] - m); sum += v[i]; }
#pragma unroll
    for (int o = 16; o > 0; o >>= 1) sum += __shfl_xor(sum, o, 32);
    float inv = 1.0f / sum;
    bf16* orow = p + (size_t)wid * 512;
#pragma unroll
    for (int i = 0; i < 16; i++) orow[lane + 32 * i] = (__bf16)(v[i] * inv);
}

// ---------------------------------------------------------------------------
// Batched 512x512 bf16 transpose (per (b,h)): dst[z][d][m] = src[b,h][m][d]
// ---------------------------------------------------------------------------
__global__ void transpose_bf16(const bf16* __restrict__ src, long sSB, long sSH,
                               int lds_, bf16* __restrict__ dst, long dStride, int ldd) {
    __shared__ bf16 tile[32][33];
    int z = blockIdx.z, bb = z >> 3, hh = z & 7;
    src += (long)bb * sSB + (long)hh * sSH;
    dst += (long)z * dStride;
    int x  = blockIdx.x * 32 + threadIdx.x;   // src col
    int y0 = blockIdx.y * 32;                 // src row base
    for (int j = threadIdx.y; j < 32; j += 8)
        tile[j][threadIdx.x] = src[(size_t)(y0 + j) * lds_ + x];
    __syncthreads();
    int xo  = blockIdx.y * 32 + threadIdx.x;  // dst col = src row
    int yo0 = blockIdx.x * 32;                // dst row base = src col
    for (int j = threadIdx.y; j < 32; j += 8)
        dst[(size_t)(yo0 + j) * ldd + xo] = tile[threadIdx.x][j];
}

// ---------------------------------------------------------------------------
// WMMA bf16 GEMM:  D[M,N] = A[M,K] * W[N,K]^T  (+ epilogue), batched over z
//   z = b*HEADS + h ; each operand offset by b*StrideB + h*StrideH
//   Block tile 128x128, K-step 32, 8 wave32 waves, wave tile 32x64.
//   Mainloop: double-buffered async global->LDS DMA (ASYNCcnt pipelined).
// ---------------------------------------------------------------------------
template <int MODE>
__global__ __launch_bounds__(256) void gemm_wmma_bf16(
    const bf16* __restrict__ A, long aSB, long aSH, int lda,
    const bf16* __restrict__ W, long wSB, long wSH, int ldw,
    void* __restrict__ C, long cSB, long cSH, int ldc,
    const float* __restrict__ bias,
    const float* __restrict__ res, int ldres,
    int K, float scale) {
    __shared__ bf16 As[2][128 * 40];
    __shared__ bf16 Bs[2][128 * 40];

    int z = blockIdx.z, bb = z >> 3, hh = z & 7;
    A += (long)bb * aSB + (long)hh * aSH;
    W += (long)bb * wSB + (long)hh * wSH;
    long coff = (long)bb * cSB + (long)hh * cSH;

    int tid = threadIdx.x, lane = tid & 31, wave = tid >> 5;
    int waveM = wave >> 1;          // 0..3 (rows of 32)
    int waveN = wave & 1;           // 0..1 (cols of 64)
    int m0 = blockIdx.y * 128, n0 = blockIdx.x * 128;

    f32x8 acc[2][4];
#pragma unroll
    for (int mi = 0; mi < 2; mi++)
#pragma unroll
        for (int ni = 0; ni < 4; ni++)
#pragma unroll
            for (int e = 0; e < 8; e++) acc[mi][ni][e] = 0.f;

    // cooperative tile transfers: 256 threads, each moves 16 halves of one row
    int ldRow = tid >> 1, ldSeg = (tid & 1) * 16;
    const bf16* aRow = A + (size_t)(m0 + ldRow) * lda + ldSeg;
    const bf16* wRow = W + (size_t)(n0 + ldRow) * ldw + ldSeg;
    int ldsOff = ldRow * 40 + ldSeg;

    // fragment addressing (A and B: interleaved K chunks)
    int ar  = lane & 15;
    int aks = (lane >> 4) * 8;       // 0 or 8
    int bc  = lane & 15;
    int bks = (lane >> 4) * 8;       // 0 or 8
    int aBase = (waveM * 32 + ar) * 40 + aks;
    int bBase = (waveN * 64 + bc) * 40 + bks;

    int nsteps = K >> 5;

#if USE_ASYNC_LDS
    // ---- double-buffered async DMA pipeline ----
    {
        // prologue: stages 0 and 1 in flight (4 asyncs each per thread)
        async_cp16(aRow,      &As[0][ldsOff]);
        async_cp16(aRow + 8,  &As[0][ldsOff + 8]);
        async_cp16(wRow,      &Bs[0][ldsOff]);
        async_cp16(wRow + 8,  &Bs[0][ldsOff + 8]);
        async_cp16(aRow + 32, &As[1][ldsOff]);
        async_cp16(aRow + 40, &As[1][ldsOff + 8]);
        async_cp16(wRow + 32, &Bs[1][ldsOff]);
        async_cp16(wRow + 40, &Bs[1][ldsOff + 8]);
    }
    for (int s = 0; s < nsteps; s++) {
        int cur = s & 1;
        // async completes in order: <=4 outstanding => stage s has landed
        if (s + 1 < nsteps) __builtin_amdgcn_s_wait_asynccnt(4);
        else                __builtin_amdgcn_s_wait_asynccnt(0);
        __syncthreads();

        bf16x16 af[2], bfv[4];
#pragma unroll
        for (int mi = 0; mi < 2; mi++) {
            const bf16* p = &As[cur][aBase + mi * 16 * 40];
            af[mi] = cat8(*(const bf16x8*)p, *(const bf16x8*)(p + 16));
        }
#pragma unroll
        for (int ni = 0; ni < 4; ni++) {
            const bf16* p = &Bs[cur][bBase + ni * 16 * 40];
            bfv[ni] = cat8(*(const bf16x8*)p, *(const bf16x8*)(p + 16));
        }
#pragma unroll
        for (int mi = 0; mi < 2; mi++)
#pragma unroll
            for (int ni = 0; ni < 4; ni++)
                acc[mi][ni] = __builtin_amdgcn_wmma_f32_16x16x32_bf16(
                    false, af[mi], false, bfv[ni], (short)0, acc[mi][ni], false, false);
        __syncthreads();   // all waves done reading buf[cur] before re-fill

        if (s + 2 < nsteps) {
            int k2 = (s + 2) << 5;
            async_cp16(aRow + k2,     &As[cur][ldsOff]);
            async_cp16(aRow + k2 + 8, &As[cur][ldsOff + 8]);
            async_cp16(wRow + k2,     &Bs[cur][ldsOff]);
            async_cp16(wRow + k2 + 8, &Bs[cur][ldsOff + 8]);
        }
    }
#else
    // ---- synchronous fallback (single buffer) ----
    for (int s = 0; s < nsteps; s++) {
        int k0 = s << 5;
        *(bf16x8*)(&As[0][ldsOff])     = *(const bf16x8*)(aRow + k0);
        *(bf16x8*)(&As[0][ldsOff + 8]) = *(const bf16x8*)(aRow + k0 + 8);
        *(bf16x8*)(&Bs[0][ldsOff])     = *(const bf16x8*)(wRow + k0);
        *(bf16x8*)(&Bs[0][ldsOff + 8]) = *(const bf16x8*)(wRow + k0 + 8);
        __syncthreads();
        bf16x16 af[2], bfv[4];
#pragma unroll
        for (int mi = 0; mi < 2; mi++) {
            const bf16* p = &As[0][aBase + mi * 16 * 40];
            af[mi] = cat8(*(const bf16x8*)p, *(const bf16x8*)(p + 16));
        }
#pragma unroll
        for (int ni = 0; ni < 4; ni++) {
            const bf16* p = &Bs[0][bBase + ni * 16 * 40];
            bfv[ni] = cat8(*(const bf16x8*)p, *(const bf16x8*)(p + 16));
        }
#pragma unroll
        for (int mi = 0; mi < 2; mi++)
#pragma unroll
            for (int ni = 0; ni < 4; ni++)
                acc[mi][ni] = __builtin_amdgcn_wmma_f32_16x16x32_bf16(
                    false, af[mi], false, bfv[ni], (short)0, acc[mi][ni], false, false);
        __syncthreads();
    }
#endif

    // epilogue: C tile element (row = r + 8*(lane>=16), col = lane&15)
    int hi = lane >> 4, cc = lane & 15;
#pragma unroll
    for (int mi = 0; mi < 2; mi++) {
#pragma unroll
        for (int ni = 0; ni < 4; ni++) {
            int col  = n0 + waveN * 64 + ni * 16 + cc;
            int rowb = m0 + waveM * 32 + mi * 16 + hi * 8;
#pragma unroll
            for (int r = 0; r < 8; r++) {
                long row = rowb + r;
                float v = acc[mi][ni][r];
                long idx = coff + row * (long)ldc + col;
                if (MODE == MODE_BF16) {
                    ((bf16*)C)[idx] = (__bf16)v;
                } else if (MODE == MODE_SCALE_F32) {
                    ((float*)C)[idx] = v * scale;
                } else if (MODE == MODE_BIAS_GELU_B16) {
                    ((bf16*)C)[idx] = (__bf16)gelu_exact(v + bias[col]);
                } else { // MODE_BIAS_RES_F32
                    ((float*)C)[idx] = v + bias[col] + res[row * (long)ldres + col];
                }
            }
        }
    }
}

// ---------------------------------------------------------------------------
// Host orchestration
// ---------------------------------------------------------------------------
extern "C" void kernel_launch(void* const* d_in, const int* in_sizes, int n_in,
                              void* d_out, int out_size, void* d_ws, size_t ws_size,
                              hipStream_t stream) {
    const float* x_in     = (const float*)d_in[0];
    const float* keys     = (const float*)d_in[1];
    const float* values   = (const float*)d_in[2];
    const float* ln1_w    = (const float*)d_in[3];
    const float* ln1_b    = (const float*)d_in[4];
    const float* qkv_w    = (const float*)d_in[5];
    const float* sa_out_w = (const float*)d_in[6];
    const float* sa_out_b = (const float*)d_in[7];
    const float* ln2_w    = (const float*)d_in[8];
    const float* ln2_b    = (const float*)d_in[9];
    const float* ff1_w1   = (const float*)d_in[10];
    const float* ff1_b1   = (const float*)d_in[11];
    const float* ff1_w2   = (const float*)d_in[12];
    const float* ff1_b2   = (const float*)d_in[13];
    const float* ln3_w    = (const float*)d_in[14];
    const float* ln3_b    = (const float*)d_in[15];
    const float* q_w      = (const float*)d_in[16];
    const float* k_w      = (const float*)d_in[17];
    const float* v_w      = (const float*)d_in[18];
    const float* ca_out_w = (const float*)d_in[19];
    const float* ca_out_b = (const float*)d_in[20];
    const float* ln4_w    = (const float*)d_in[21];
    const float* ln4_b    = (const float*)d_in[22];
    const float* ff2_w1   = (const float*)d_in[23];
    const float* ff2_b1   = (const float*)d_in[24];
    const float* ff2_w2   = (const float*)d_in[25];
    const float* ff2_b2   = (const float*)d_in[26];

    // workspace carve-up (256B aligned)
    char* base = (char*)d_ws;
    size_t off = 0;
    auto alloc = [&](size_t bytes) -> char* {
        char* p = base + off;
        off = (off + bytes + 255) & ~(size_t)255;
        return p;
    };
    float* xbuf   = (float*)alloc((size_t)TOK_ * DIM_ * 4);
    bf16*  lnbuf  = (bf16*) alloc((size_t)TOK_ * DIM_ * 2);
    bf16*  qkvbuf = (bf16*) alloc((size_t)TOK_ * 3 * INNER_ * 2);
    float* scores = (float*)alloc((size_t)B_ * HEADS_ * N_ * S_ * 4);
    bf16*  probs  = (bf16*) alloc((size_t)B_ * HEADS_ * N_ * S_ * 2);
    bf16*  vt     = (bf16*) alloc((size_t)B_ * HEADS_ * S_ * DHEAD_ * 2);
    bf16*  attno  = (bf16*) alloc((size_t)TOK_ * INNER_ * 2);
    bf16*  ffbuf  = (bf16*) alloc((size_t)TOK_ * MLP_ * 2);
    bf16*  wstage = (bf16*) alloc((size_t)3 * INNER_ * DIM_ * 2);
    bf16*  keysbf = (bf16*) alloc((size_t)TOK_ * DIM_ * 2);
    bf16*  valsbf = (bf16*) alloc((size_t)TOK_ * DIM_ * 2);

    auto cvt = [&](const float* s, bf16* d, size_t n) {
        long n4 = (long)(n / 4);
        int blocks = (int)((n4 + 255) / 256);
        if (blocks > 4096) blocks = 4096;
        cvt_f32_bf16<<<blocks, 256, 0, stream>>>((const float4*)s, d, n4);
    };
    auto lnorm = [&](const float* w, const float* b) {
        layernorm_bf16<<<TOK_ / 8, 256, 0, stream>>>(xbuf, w, b, lnbuf, TOK_);
    };
    auto gemm = [&](const bf16* A, long aSB, long aSH, int lda,
                    const bf16* W, long wSB, long wSH, int ldw,
                    void* C, long cSB, long cSH, int ldc,
                    const float* bias, const float* res, int ldres,
                    int M, int Nn, int K, int Z, int mode, float scl) {
        dim3 g(Nn / 128, M / 128, Z);
        switch (mode) {
        case MODE_BF16:
            gemm_wmma_bf16<MODE_BF16><<<g, 256, 0, stream>>>(
                A, aSB, aSH, lda, W, wSB, wSH, ldw, C, cSB, cSH, ldc, bias, res, ldres, K, scl);
            break;
        case MODE_SCALE_F32:
            gemm_wmma_bf16<MODE_SCALE_F32><<<g, 256, 0, stream>>>(
                A, aSB, aSH, lda, W, wSB, wSH, ldw, C, cSB, cSH, ldc, bias, res, ldres, K, scl);
            break;
        case MODE_BIAS_GELU_B16:
            gemm_wmma_bf16<MODE_BIAS_GELU_B16><<<g, 256, 0, stream>>>(
                A, aSB, aSH, lda, W, wSB, wSH, ldw, C, cSB, cSH, ldc, bias, res, ldres, K, scl);
            break;
        default:
            gemm_wmma_bf16<MODE_BIAS_RES_F32><<<g, 256, 0, stream>>>(
                A, aSB, aSH, lda, W, wSB, wSH, ldw, C, cSB, cSH, ldc, bias, res, ldres, K, scl);
            break;
        }
    };
    auto softmax = [&]() {
        softmax_bf16<<<(B_ * HEADS_ * N_) / 8, 256, 0, stream>>>(scores, probs, B_ * HEADS_ * N_);
    };
    auto transp = [&](const bf16* src, long sSB, long sSH, int lds_) {
        dim3 g(512 / 32, 512 / 32, B_ * HEADS_);
        dim3 blk(32, 8);
        transpose_bf16<<<g, blk, 0, stream>>>(src, sSB, sSH, lds_, vt, 512L * 512, 512);
    };

    const long ZHS = (long)HEADS_ * N_ * S_;       // 2097152 (per-b stride, scores/probs)
    const long ZS  = (long)N_ * S_;                // 262144  (per-h stride)

    // init residual stream + constant bf16 keys/values
    (void)hipMemcpyAsync(xbuf, x_in, (size_t)TOK_ * DIM_ * 4, hipMemcpyDeviceToDevice, stream);
    cvt(keys,   keysbf, (size_t)TOK_ * DIM_);
    cvt(values, valsbf, (size_t)TOK_ * DIM_);

    for (int i = 0; i < 4; i++) {
        // ---------------- self-attention ----------------
        lnorm(ln1_w + i * DIM_, ln1_b + i * DIM_);
        cvt(qkv_w + (size_t)i * 3 * INNER_ * DIM_, wstage, (size_t)3 * INNER_ * DIM_);
        gemm(lnbuf, 0, 0, DIM_, wstage, 0, 0, DIM_, qkvbuf, 0, 0, 3 * INNER_,
             nullptr, nullptr, 0, TOK_, 3 * INNER_, DIM_, 1, MODE_BF16, 0.f);
        // scores = Q K^T * scale  (head h = col slice h*512; row stride 12288)
        gemm(qkvbuf,          (long)N_ * 3 * INNER_, DHEAD_, 3 * INNER_,
             qkvbuf + INNER_, (long)S_ * 3 * INNER_, DHEAD_, 3 * INNER_,
             scores, ZHS, ZS, S_, nullptr, nullptr, 0,
             N_, S_, DHEAD_, B_ * HEADS_, MODE_SCALE_F32, SCALE_);
        softmax();
        transp(qkvbuf + 2 * INNER_, (long)S_ * 3 * INNER_, DHEAD_, 3 * INNER_);
        // O = P V  -> merge heads into attno[b, n, h*512 + d]
        gemm(probs, ZHS, ZS, S_, vt, ZHS, ZS, S_,
             attno, (long)N_ * INNER_, DHEAD_, INNER_, nullptr, nullptr, 0,
             N_, DHEAD_, S_, B_ * HEADS_, MODE_BF16, 0.f);
        cvt(sa_out_w + (size_t)i * DIM_ * INNER_, wstage, (size_t)DIM_ * INNER_);
        gemm(attno, 0, 0, INNER_, wstage, 0, 0, INNER_, xbuf, 0, 0, DIM_,
             sa_out_b + i * DIM_, xbuf, DIM_, TOK_, DIM_, INNER_, 1, MODE_BIAS_RES_F32, 0.f);

        // ---------------- ffn1 ----------------
        lnorm(ln2_w + i * DIM_, ln2_b + i * DIM_);
        cvt(ff1_w1 + (size_t)i * MLP_ * DIM_, wstage, (size_t)MLP_ * DIM_);
        gemm(lnbuf, 0, 0, DIM_, wstage, 0, 0, DIM_, ffbuf, 0, 0, MLP_,
             ff1_b1 + i * MLP_, nullptr, 0, TOK_, MLP_, DIM_, 1, MODE_BIAS_GELU_B16, 0.f);
        cvt(ff1_w2 + (size_t)i * DIM_ * MLP_, wstage, (size_t)DIM_ * MLP_);
        gemm(ffbuf, 0, 0, MLP_, wstage, 0, 0, MLP_, xbuf, 0, 0, DIM_,
             ff1_b2 + i * DIM_, xbuf, DIM_, TOK_, DIM_, MLP_, 1, MODE_BIAS_RES_F32, 0.f);

        // ---------------- cross-attention ----------------
        lnorm(ln3_w + i * DIM_, ln3_b + i * DIM_);
        bf16* qbuf = qkvbuf;
        bf16* kbuf = qkvbuf + (size_t)TOK_ * INNER_;
        bf16* vbuf = kbuf  + (size_t)TOK_ * INNER_;
        cvt(q_w + (size_t)i * INNER_ * DIM_, wstage, (size_t)INNER_ * DIM_);
        gemm(lnbuf, 0, 0, DIM_, wstage, 0, 0, DIM_, qbuf, 0, 0, INNER_,
             nullptr, nullptr, 0, TOK_, INNER_, DIM_, 1, MODE_BF16, 0.f);
        cvt(k_w + (size_t)i * INNER_ * DIM_, wstage, (size_t)INNER_ * DIM_);
        gemm(keysbf, 0, 0, DIM_, wstage, 0, 0, DIM_, kbuf, 0, 0, INNER_,
             nullptr, nullptr, 0, TOK_, INNER_, DIM_, 1, MODE_BF16, 0.f);
        cvt(v_w + (size_t)i * INNER_ * DIM_, wstage, (size_t)INNER_ * DIM_);
        gemm(valsbf, 0, 0, DIM_, wstage, 0, 0, DIM_, vbuf, 0, 0, INNER_,
             nullptr, nullptr, 0, TOK_, INNER_, DIM_, 1, MODE_BF16, 0.f);
        // raw-reshape quirk: head h of batch b = contiguous [512,512] chunk at
        //   Q[b] + h*262144  (b-stride 2097152), row stride 512
        gemm(qbuf, ZHS, ZS, DHEAD_, kbuf, ZHS, ZS, DHEAD_,
             scores, ZHS, ZS, S_, nullptr, nullptr, 0,
             N_, S_, DHEAD_, B_ * HEADS_, MODE_SCALE_F32, SCALE_);
        softmax();
        transp(vbuf, ZHS, ZS, DHEAD_);
        gemm(probs, ZHS, ZS, S_, vt, ZHS, ZS, S_,
             attno, (long)N_ * INNER_, DHEAD_, INNER_, nullptr, nullptr, 0,
             N_, DHEAD_, S_, B_ * HEADS_, MODE_BF16, 0.f);
        cvt(ca_out_w + (size_t)i * DIM_ * INNER_, wstage, (size_t)DIM_ * INNER_);
        gemm(attno, 0, 0, INNER_, wstage, 0, 0, INNER_, xbuf, 0, 0, DIM_,
             ca_out_b + i * DIM_, xbuf, DIM_, TOK_, DIM_, INNER_, 1, MODE_BIAS_RES_F32, 0.f);

        // ---------------- ffn2 ----------------
        lnorm(ln4_w + i * DIM_, ln4_b + i * DIM_);
        cvt(ff2_w1 + (size_t)i * MLP_ * DIM_, wstage, (size_t)MLP_ * DIM_);
        gemm(lnbuf, 0, 0, DIM_, wstage, 0, 0, DIM_, ffbuf, 0, 0, MLP_,
             ff2_b1 + i * MLP_, nullptr, 0, TOK_, MLP_, DIM_, 1, MODE_BIAS_GELU_B16, 0.f);
        cvt(ff2_w2 + (size_t)i * DIM_ * MLP_, wstage, (size_t)DIM_ * MLP_);
        gemm(ffbuf, 0, 0, MLP_, wstage, 0, 0, MLP_, xbuf, 0, 0, DIM_,
             ff2_b2 + i * DIM_, xbuf, DIM_, TOK_, DIM_, MLP_, 1, MODE_BIAS_RES_F32, 0.f);
    }

    (void)hipMemcpyAsync(d_out, xbuf, (size_t)TOK_ * DIM_ * 4, hipMemcpyDeviceToDevice, stream);
}
